// SS2D_my_76209899701004
// MI455X (gfx1250) — compile-verified, matrix operand order checked by reading, not run on hardware
//
#include <hip/hip_runtime.h>
#include <stdint.h>

// ---- problem constants ----
#define BQ   8
#define DQ   192
#define HQ   64
#define WQ   64
#define LQ   4096      // H*W
#define NQ   16        // d_state
#define RQ   6         // dt_rank
#define KQ   4         // scan directions
#define CQ   38        // R + 2N
#define CP   40        // padded x_dbl row stride (floats, 160B -> float4 aligned)
#define NCH  16        // chunks along L
#define LCH  256       // chunk length (NCH*LCH == LQ)

typedef __attribute__((ext_vector_type(16))) _Float16     v16h;
typedef __attribute__((ext_vector_type(8)))  float        v8f;
typedef __attribute__((ext_vector_type(4)))  unsigned int u32x4;
typedef __attribute__((ext_vector_type(8)))  int          i32x8;
typedef __attribute__((ext_vector_type(4)))  int          i32x4;

#if __has_builtin(__builtin_amdgcn_tensor_load_to_lds) && __has_builtin(__builtin_amdgcn_s_wait_tensorcnt)
#define HAVE_TDM 1
#else
#define HAVE_TDM 0
#endif

// CrossScan index: memory position (row-major in x) of traversal step l of direction k.
__device__ __forceinline__ int map_idx(int k, int l) {
  int t = (k >= 2) ? (LQ - 1 - l) : l;
  if (k & 1) t = (t & (HQ - 1)) * WQ + (t >> 6);   // column-major traversal
  return t;
}

__device__ __forceinline__ float softplus_f(float v) {
  return (v > 20.f) ? v : log1pf(__expf(v));
}

#if HAVE_TDM
// 1-D TDM copy: nelem f32 elements from global -> LDS byte offset lds_off.
__device__ __forceinline__ void tdm_load_1d(unsigned int lds_off, const void* gptr,
                                            unsigned int nelem) {
  unsigned long long ga = (unsigned long long)(uintptr_t)gptr;
  // D# group0: count=1, gather off, lds_addr, 57-bit global addr, type=2 ("image")
  u32x4 g0 = { 1u,
               lds_off,
               (unsigned int)(ga & 0xFFFFFFFFu),
               (unsigned int)((ga >> 32) & 0x1FFFFFFu) | (2u << 30) };
  // D# group1: data_size=4B, tensor_dim0=nelem, tensor_dim1=1, tile_dim0=nelem,
  // tile_dim1=1, no padding/iterate/multicast, stride0=nelem.
  i32x8 g1 = { (int)(2u << 16),
               (int)((nelem & 0xFFFFu) << 16),
               (int)(((nelem >> 16) & 0xFFFFu) | (1u << 16)),
               (int)((nelem & 0xFFFFu) << 16),
               1,
               (int)nelem,
               0, 0 };
  i32x4 gz = { 0, 0, 0, 0 };
#if defined(__clang_major__) && (__clang_major__ >= 23)
  i32x8 gz8 = { 0, 0, 0, 0, 0, 0, 0, 0 };
  __builtin_amdgcn_tensor_load_to_lds(g0, g1, gz, gz, gz8, 0);
#else
  __builtin_amdgcn_tensor_load_to_lds(g0, g1, gz, gz, 0);
#endif
  __builtin_amdgcn_s_wait_tensorcnt(0);
}
#endif

// Stage this (b,k,chunk)'s x_dbl slab [LCH x CP] into LDS (6 waves, disjoint slices).
__device__ __forceinline__ void stage_slab(float* slab, const float* __restrict__ xdbl,
                                           int b, int k, int c, int wv) {
  const size_t rowbase = ((size_t)(b * KQ + k) * LQ + (size_t)c * LCH);
#if HAVE_TDM
  int st = (wv * LCH) / 6;
  int en = ((wv + 1) * LCH) / 6;
  unsigned int nel = (unsigned int)(en - st) * CP;
  unsigned int ldsb = (unsigned int)(uintptr_t)(&slab[0]) + (unsigned int)st * CP * 4u;
  tdm_load_1d(ldsb, xdbl + (rowbase + st) * CP, nel);
#else
  for (int i = threadIdx.x; i < LCH * CP; i += 192)
    slab[i] = xdbl[rowbase * CP + i];
#endif
  __syncthreads();
}

// ---------------------------------------------------------------------------
// Kernel 1: x_dbl[b,k,l,c] = sum_d W[k,c,d] * xs[b,k,d,l]   (WMMA f16->f32)
// One wave per 16-row l-tile; 3 c-tiles (48 cols, 38 valid); K-dim 192 = 6x32.
// ---------------------------------------------------------------------------
__global__ __launch_bounds__(256) void k_xdbl_wmma(const float* __restrict__ x,
                                                   const float* __restrict__ Wp,
                                                   float* __restrict__ xdbl) {
  const int lane = threadIdx.x & 31;
  const int wv   = threadIdx.x >> 5;
  const int gw   = blockIdx.x * 8 + wv;        // global wave = tile id
  const int ltile = gw & 255;                  // L/16 = 256 tiles
  const int k     = (gw >> 8) & 3;
  const int b     = gw >> 10;
  const int l0    = ltile << 4;
  const int half  = lane >> 4;                 // K-half of the fragment
  const int mrow  = lane & 15;                 // tile row (A) / tile col (B,D)
  const int lidx  = map_idx(k, l0 + mrow);     // CrossScan folded into the gather

  const float* xb = x + (size_t)b * DQ * LQ;
  const float* wk = Wp + (size_t)k * CQ * DQ;

  v8f acc[3] = { {}, {}, {} };

  for (int kk = 0; kk < 6; ++kk) {
    const int db = kk * 32 + half * 8;
    v16h a;
#pragma unroll
    for (int j = 0; j < 8; ++j) {
      a[j]     = (_Float16)xb[(size_t)(db + j) * LQ + lidx];
      a[j + 8] = (_Float16)xb[(size_t)(db + 16 + j) * LQ + lidx];
    }
#pragma unroll
    for (int ct = 0; ct < 3; ++ct) {
      const int c = ct * 16 + mrow;
      v16h bf;
      if (c < CQ) {
        const float* wr = wk + (size_t)c * DQ;
#pragma unroll
        for (int j = 0; j < 8; ++j) {
          bf[j]     = (_Float16)wr[db + j];
          bf[j + 8] = (_Float16)wr[db + 16 + j];
        }
      } else {
#pragma unroll
        for (int j = 0; j < 16; ++j) bf[j] = (_Float16)0.f;
      }
      acc[ct] = __builtin_amdgcn_wmma_f32_16x16x32_f16(
          false, a, false, bf, (short)0, acc[ct], false, false);
    }
  }

  float* outb = xdbl + ((size_t)(b * KQ + k) * LQ + l0 + half * 8) * CP;
#pragma unroll
  for (int ct = 0; ct < 3; ++ct) {
    const int c = ct * 16 + mrow;
    if (c < CQ) {
#pragma unroll
      for (int r = 0; r < 8; ++r) outb[(size_t)r * CP + c] = acc[ct][r];
    }
  }
}

// ---------------------------------------------------------------------------
// Kernel 2a: chunk-local scan with h0=0. Emits per-(b,k,chunk,d): sum(delta)
// and local end state h[16]. Block = (b,k,chunk), 6 waves over D=192.
// ---------------------------------------------------------------------------
__global__ __launch_bounds__(192) void k_scan_passA(const float* __restrict__ x,
                                                    const float* __restrict__ xdbl,
                                                    const float* __restrict__ dtw_g,
                                                    const float* __restrict__ dtb_g,
                                                    const float* __restrict__ Alog,
                                                    float* __restrict__ hend,
                                                    float* __restrict__ dsums) {
  const int lane = threadIdx.x & 31, wv = threadIdx.x >> 5;
  const int c = blockIdx.x & (NCH - 1);
  const int k = (blockIdx.x >> 4) & 3;
  const int b = blockIdx.x >> 6;
  const int d = wv * 32 + lane;

  __shared__ __align__(16) float slab[LCH * CP];
  stage_slab(slab, xdbl, b, k, c, wv);

  float dtw[RQ];
#pragma unroll
  for (int r = 0; r < RQ; ++r) dtw[r] = dtw_g[(k * DQ + d) * RQ + r];
  const float bias = dtb_g[k * DQ + d];
  float Aa[NQ];
#pragma unroll
  for (int n = 0; n < NQ; ++n) Aa[n] = -__expf(Alog[(k * DQ + d) * NQ + n]);
  const float* px = x + (size_t)(b * DQ + d) * LQ;

  float h[NQ];
#pragma unroll
  for (int n = 0; n < NQ; ++n) h[n] = 0.f;
  float ds = 0.f;
  const int l0 = c * LCH;

  for (int t = 0; t < LCH; ++t) {
    const float4* rw = (const float4*)(slab + t * CP);
    float rloc[24];
#pragma unroll
    for (int i = 0; i < 6; ++i) {
      float4 q = rw[i];
      rloc[4 * i] = q.x; rloc[4 * i + 1] = q.y; rloc[4 * i + 2] = q.z; rloc[4 * i + 3] = q.w;
    }
    float dt = bias;
#pragma unroll
    for (int r = 0; r < RQ; ++r) dt = fmaf(dtw[r], rloc[r], dt);
    const float delta = softplus_f(dt);
    ds += delta;
    const float u  = px[map_idx(k, l0 + t)];
    const float du = delta * u;
#pragma unroll
    for (int n = 0; n < NQ; ++n) {
      const float e = __expf(delta * Aa[n]);
      h[n] = fmaf(e, h[n], du * rloc[6 + n]);
    }
  }

  const size_t base = (((size_t)(b * KQ + k) * NCH + c) * DQ + d);
  dsums[base] = ds;
#pragma unroll
  for (int n = 0; n < NQ; ++n) hend[base * NQ + n] = h[n];
}

// ---------------------------------------------------------------------------
// Kernel 2b: serial chunk stitch.  Hin[c] = exp(A*dsum[c-1])*Hin[c-1] + hend[c-1]
// One thread per (b,k,d,n).
// ---------------------------------------------------------------------------
__global__ __launch_bounds__(256) void k_combine(const float* __restrict__ Alog,
                                                 const float* __restrict__ dsums,
                                                 const float* __restrict__ hend,
                                                 float* __restrict__ hin) {
  const int tid = blockIdx.x * 256 + threadIdx.x;   // < B*K*D*N
  const int n  = tid & 15;
  const int d  = (tid >> 4) % DQ;
  const int kb = tid / (16 * DQ);
  const int k  = kb & 3;
  const int b  = kb >> 2;
  const float Av = -__expf(Alog[(k * DQ + d) * NQ + n]);
  float h = 0.f;
  for (int c = 0; c < NCH; ++c) {
    const size_t base = (((size_t)(b * KQ + k) * NCH + c) * DQ + d);
    hin[base * NQ + n] = h;
    h = fmaf(__expf(Av * dsums[base]), h, hend[base * NQ + n]);
  }
}

// ---------------------------------------------------------------------------
// Kernel 2c: replay each chunk from its true initial state; emit
// ys[b,k,l,d] = C_t.h_t + Ds*u  (lane-coalesced in d).
// ---------------------------------------------------------------------------
__global__ __launch_bounds__(192) void k_scan_passC(const float* __restrict__ x,
                                                    const float* __restrict__ xdbl,
                                                    const float* __restrict__ dtw_g,
                                                    const float* __restrict__ dtb_g,
                                                    const float* __restrict__ Alog,
                                                    const float* __restrict__ Dsg,
                                                    const float* __restrict__ hin,
                                                    float* __restrict__ ys) {
  const int lane = threadIdx.x & 31, wv = threadIdx.x >> 5;
  const int c = blockIdx.x & (NCH - 1);
  const int k = (blockIdx.x >> 4) & 3;
  const int b = blockIdx.x >> 6;
  const int d = wv * 32 + lane;

  __shared__ __align__(16) float slab[LCH * CP];
  stage_slab(slab, xdbl, b, k, c, wv);

  float dtw[RQ];
#pragma unroll
  for (int r = 0; r < RQ; ++r) dtw[r] = dtw_g[(k * DQ + d) * RQ + r];
  const float bias = dtb_g[k * DQ + d];
  float Aa[NQ];
#pragma unroll
  for (int n = 0; n < NQ; ++n) Aa[n] = -__expf(Alog[(k * DQ + d) * NQ + n]);
  const float Dsv = Dsg[k * DQ + d];
  const float* px = x + (size_t)(b * DQ + d) * LQ;

  const size_t base = (((size_t)(b * KQ + k) * NCH + c) * DQ + d);
  float h[NQ];
#pragma unroll
  for (int n = 0; n < NQ; ++n) h[n] = hin[base * NQ + n];

  const int l0 = c * LCH;
  float* ysb = ys + ((size_t)(b * KQ + k) * LQ) * DQ + d;

  for (int t = 0; t < LCH; ++t) {
    const float4* rw = (const float4*)(slab + t * CP);
    float rloc[40];
#pragma unroll
    for (int i = 0; i < 10; ++i) {
      float4 q = rw[i];
      rloc[4 * i] = q.x; rloc[4 * i + 1] = q.y; rloc[4 * i + 2] = q.z; rloc[4 * i + 3] = q.w;
    }
    float dt = bias;
#pragma unroll
    for (int r = 0; r < RQ; ++r) dt = fmaf(dtw[r], rloc[r], dt);
    const float delta = softplus_f(dt);
    const float u  = px[map_idx(k, l0 + t)];
    const float du = delta * u;
    float y = 0.f;
#pragma unroll
    for (int n = 0; n < NQ; ++n) {
      const float e = __expf(delta * Aa[n]);
      h[n] = fmaf(e, h[n], du * rloc[6 + n]);
      y = fmaf(rloc[22 + n], h[n], y);
    }
    ysb[(size_t)(l0 + t) * DQ] = fmaf(Dsv, u, y);
  }
}

// ---------------------------------------------------------------------------
// Kernel 3: CrossMerge + channel LayerNorm. One wave per (b, pixel):
// lanes stride over 192 channels, shfl_xor wave32 reduction for mean/var.
// ---------------------------------------------------------------------------
__global__ __launch_bounds__(256) void k_merge_ln(const float* __restrict__ ys,
                                                  const float* __restrict__ lng,
                                                  const float* __restrict__ lnb,
                                                  float* __restrict__ out) {
  const int lane = threadIdx.x & 31, wv = threadIdx.x >> 5;
  const int pos = blockIdx.x * 8 + wv;     // B*L waves total
  const int b = pos >> 12;                 // L = 4096
  const int l = pos & (LQ - 1);
  const int hh = l >> 6, ww = l & 63;
  const int l1 = ww * HQ + hh;             // transposed-traversal step for this pixel

  const size_t r0 = ((size_t)(b * KQ + 0) * LQ + l) * DQ;
  const size_t r1 = ((size_t)(b * KQ + 1) * LQ + l1) * DQ;
  const size_t r2 = ((size_t)(b * KQ + 2) * LQ + (LQ - 1 - l)) * DQ;
  const size_t r3 = ((size_t)(b * KQ + 3) * LQ + (LQ - 1 - l1)) * DQ;

  float v[6], s1 = 0.f, s2 = 0.f;
#pragma unroll
  for (int j = 0; j < 6; ++j) {
    const int d = lane + j * 32;
    const float t = ys[r0 + d] + ys[r1 + d] + ys[r2 + d] + ys[r3 + d];
    v[j] = t; s1 += t; s2 += t * t;
  }
#pragma unroll
  for (int off = 16; off > 0; off >>= 1) {
    s1 += __shfl_xor(s1, off, 32);
    s2 += __shfl_xor(s2, off, 32);
  }
  const float mean = s1 * (1.f / DQ);
  const float var  = s2 * (1.f / DQ) - mean * mean;
  const float inv  = rsqrtf(var + 1e-5f);
#pragma unroll
  for (int j = 0; j < 6; ++j) {
    const int d = lane + j * 32;
    out[((size_t)(b * DQ + d)) * LQ + l] = (v[j] - mean) * inv * lng[d] + lnb[d];
  }
}

// ---------------------------------------------------------------------------
extern "C" void kernel_launch(void* const* d_in, const int* in_sizes, int n_in,
                              void* d_out, int out_size, void* d_ws, size_t ws_size,
                              hipStream_t stream) {
  (void)in_sizes; (void)n_in; (void)out_size; (void)ws_size;
  const float* x    = (const float*)d_in[0];
  const float* xpw  = (const float*)d_in[1];
  const float* dtw  = (const float*)d_in[2];
  const float* dtb  = (const float*)d_in[3];
  const float* Alog = (const float*)d_in[4];
  const float* Dsg  = (const float*)d_in[5];
  const float* lng  = (const float*)d_in[6];
  const float* lnb  = (const float*)d_in[7];

  float* ws = (float*)d_ws;
  const size_t n_xdbl = (size_t)BQ * KQ * LQ * CP;       // ~21 MB
  const size_t n_ys   = (size_t)BQ * KQ * LQ * DQ;       // ~101 MB
  const size_t n_dsum = (size_t)BQ * KQ * NCH * DQ;
  const size_t n_hend = n_dsum * NQ;                     // ~6.3 MB each
  float* xdbl  = ws;
  float* ysb   = xdbl + n_xdbl;
  float* dsums = ysb + n_ys;
  float* hend  = dsums + n_dsum;
  float* hin   = hend + n_hend;
  float* out   = (float*)d_out;

  k_xdbl_wmma <<<1024, 256, 0, stream>>>(x, xpw, xdbl);
  k_scan_passA<<<BQ * KQ * NCH, 192, 0, stream>>>(x, xdbl, dtw, dtb, Alog, hend, dsums);
  k_combine   <<<(BQ * KQ * DQ * NQ) / 256, 256, 0, stream>>>(Alog, dsums, hend, hin);
  k_scan_passC<<<BQ * KQ * NCH, 192, 0, stream>>>(x, xdbl, dtw, dtb, Alog, Dsg, hin, ysb);
  k_merge_ln  <<<BQ * LQ / 8, 256, 0, stream>>>(ysb, lng, lnb, out);
}
